// QuantizedAttention_56066503082631
// MI455X (gfx1250) — compile-verified, matrix operand order checked by reading, not run on hardware
//
#include <hip/hip_runtime.h>

// ---------------------------------------------------------------------------
// QuantizedAttention for MI455X (gfx1250, wave32, WMMA 16x16x32 bf16,
// async global->LDS tile staging, double-buffered GEMM mainloop)
// ---------------------------------------------------------------------------

typedef __bf16 bf16_t;
typedef __bf16 v8bf  __attribute__((ext_vector_type(8)));
typedef __bf16 v16bf __attribute__((ext_vector_type(16)));
typedef float  v8f   __attribute__((ext_vector_type(8)));

#define NUM_B 8
#define NUM_N 1024
#define NUM_C 768
#define NUM_H 12
#define DIM_D 64
#define TOPK  32
#define LDA_S 40   // bf16 elems per LDS row: 80B (multiple of 16B, bank-skewed)

// sign(t) * 2^floor(log2|t|): clear mantissa bits of f32 (exact in bf16)
__device__ __forceinline__ float exq(float t) {
    unsigned u = __float_as_uint(t);
    return __uint_as_float(u & 0xFF800000u);
}

__device__ __forceinline__ v16bf cat16(v8bf a, v8bf b) {
    return __builtin_shufflevector(a, b, 0,1,2,3,4,5,6,7,8,9,10,11,12,13,14,15);
}

__device__ __forceinline__ v8f wmma_bf(v16bf a, v16bf b, v8f c) {
    return __builtin_amdgcn_wmma_f32_16x16x32_bf16(
        false, a, false, b, (short)0, c, false, false);
}

// generic LDS pointer -> workgroup-relative LDS byte offset (addr[31:0])
__device__ __forceinline__ unsigned lds_off(const void* p) {
    return (unsigned)(unsigned long long)p;
}

// CDNA5 async copy: 16 bytes global -> LDS, tracked by ASYNCcnt (per lane)
__device__ __forceinline__ void async_copy_b128(unsigned lds_byte_off, const void* gaddr) {
    asm volatile("global_load_async_to_lds_b128 %0, %1, off"
                 :: "v"(lds_byte_off), "v"((unsigned long long)gaddr)
                 : "memory");
}

__device__ __forceinline__ void wait_asynccnt0() {
    asm volatile("s_wait_asynccnt 0x0" ::: "memory");
}

// ---------------------------------------------------------------------------
// f32 -> bf16 bulk convert (8 elems / thread)
// ---------------------------------------------------------------------------
__global__ __launch_bounds__(256) void convert_bf16_kernel(
    const float* __restrict__ src, bf16_t* __restrict__ dst, int n8)
{
    int i = blockIdx.x * blockDim.x + threadIdx.x;
    if (i >= n8) return;
    const float4* s = reinterpret_cast<const float4*>(src) + (size_t)i * 2;
    float4 a = s[0], b = s[1];
    v8bf v;
    v[0] = (bf16_t)a.x; v[1] = (bf16_t)a.y; v[2] = (bf16_t)a.z; v[3] = (bf16_t)a.w;
    v[4] = (bf16_t)b.x; v[5] = (bf16_t)b.y; v[6] = (bf16_t)b.z; v[7] = (bf16_t)b.w;
    reinterpret_cast<v8bf*>(dst)[i] = v;
}

// ---------------------------------------------------------------------------
// GEMM:  out[M][Ncols] = A[M][K] @ Bw[Ncols][K]^T + bias   (A, Bw in bf16)
// Block tile 64x128, 256 threads = 8 waves (4 row x 2 col), wave = 16x64.
// K chunk = 32, staged via global_load_async_to_lds_b128, double buffered.
// MODE 0: epilogue scatters QKV -> [B,H,N,d] q/eq/k/ek (bf16), v (f32)
// MODE 1: epilogue writes f32 out + bias
// ---------------------------------------------------------------------------
template <int MODE>
__global__ __launch_bounds__(256) void gemm_kernel(
    const bf16_t* __restrict__ Abf, const bf16_t* __restrict__ Bw,
    const float*  __restrict__ bias,
    int Kdim, int Ncols,
    float* __restrict__ outf,
    bf16_t* __restrict__ qbf, bf16_t* __restrict__ eqbf,
    bf16_t* __restrict__ kbf, bf16_t* __restrict__ ekbf,
    float* __restrict__ vf)
{
    __shared__ __align__(16) bf16_t As[2][64  * LDA_S];
    __shared__ __align__(16) bf16_t Bs[2][128 * LDA_S];

    const int tid  = threadIdx.x;
    const int lane = tid & 31;
    const int wv   = tid >> 5;
    const int wRow = wv & 3;
    const int wCol = wv >> 2;
    const int rowBase = blockIdx.y * 64;
    const int colBase = blockIdx.x * 128;

    // staging geometry: A 64x32 bf16 = 256x16B (1/thread); B 128x32 = 512x16B (2/thread)
    const int aRow = tid >> 2, aSeg = (tid & 3) * 8;
    const int bRow = tid >> 1, bSeg = (tid & 1) * 16;
    const bf16_t* aSrcBase = Abf + (size_t)(rowBase + aRow) * Kdim + aSeg;
    const bf16_t* bSrcBase = Bw  + (size_t)(colBase + bRow) * Kdim + bSeg;

    auto stage = [&](int buf, int k0) {
        async_copy_b128(lds_off(&As[buf][aRow * LDA_S + aSeg]),     aSrcBase + k0);
        async_copy_b128(lds_off(&Bs[buf][bRow * LDA_S + bSeg]),     bSrcBase + k0);
        async_copy_b128(lds_off(&Bs[buf][bRow * LDA_S + bSeg + 8]), bSrcBase + k0 + 8);
    };

    v8f acc[4] = {};
    const int nk = Kdim >> 5;
    stage(0, 0);

    for (int kc = 0; kc < nk; ++kc) {
        const int cur = kc & 1;
        wait_asynccnt0();
        __syncthreads();                         // buf[cur] visible to all waves
        if (kc + 1 < nk) stage(cur ^ 1, (kc + 1) << 5);   // overlap with WMMAs below

        const int off = (lane < 16) ? 0 : 8;
        const int ar  = wRow * 16 + (lane & 15);
        const bf16_t* asb = &As[cur][ar * LDA_S];
        v16bf afrag = cat16(*reinterpret_cast<const v8bf*>(asb + off),
                            *reinterpret_cast<const v8bf*>(asb + off + 16));
        #pragma unroll
        for (int s = 0; s < 4; ++s) {
            const int bc = wCol * 64 + s * 16 + (lane & 15);
            const bf16_t* bsb = &Bs[cur][bc * LDA_S];
            v16bf bfrag = cat16(*reinterpret_cast<const v8bf*>(bsb + off),
                                *reinterpret_cast<const v8bf*>(bsb + off + 16));
            acc[s] = wmma_bf(afrag, bfrag, acc[s]);
        }
        __syncthreads();                         // reads of buf[cur] done
    }

    // ---- epilogue (C layout: VGPR i -> row i or i+8; col = lane&15)
    #pragma unroll
    for (int s = 0; s < 4; ++s) {
        const int col = colBase + wCol * 64 + s * 16 + (lane & 15);
        const float bvv = bias[col];
        #pragma unroll
        for (int i = 0; i < 8; ++i) {
            const int row = rowBase + wRow * 16 + i + ((lane < 16) ? 0 : 8);
            const float val = acc[s][i] + bvv;
            if constexpr (MODE == 1) {
                outf[(size_t)row * Ncols + col] = val;
            } else {
                const int region = col / NUM_C;            // 0=q, 1=k, 2=v
                const int cc = col - region * NUM_C;
                const int h  = cc >> 6, dd = cc & 63;
                const int b  = row >> 10, n = row & 1023;
                const size_t idx = (((size_t)b * NUM_H + h) * NUM_N + n) * DIM_D + dd;
                if (region == 0)      { qbf[idx] = (bf16_t)val; eqbf[idx] = (bf16_t)exq(val); }
                else if (region == 1) { kbf[idx] = (bf16_t)val; ekbf[idx] = (bf16_t)exq(val); }
                else                  { vf[idx]  = val; }
            }
        }
    }
}

// ---------------------------------------------------------------------------
// Fused attention: per (b, h, 16-row tile):
//   WMMA true + pred score stripes (16 x 1024) into LDS,
//   per-row iterative top-32 (register candidates + shfl reductions),
//   32-wide softmax, sparse attn @ V, write Y[B,N,C] bf16.
// ---------------------------------------------------------------------------
__global__ __launch_bounds__(256) void attn_kernel(
    const bf16_t* __restrict__ qbf, const bf16_t* __restrict__ eqbf,
    const bf16_t* __restrict__ kbf, const bf16_t* __restrict__ ekbf,
    const float*  __restrict__ vf,  bf16_t* __restrict__ ybf)
{
    extern __shared__ char smem[];
    float* s_true = (float*)smem;                          // [16][1024]
    float* s_pred = (float*)(smem + 16 * 1024 * sizeof(float));

    const int tid  = threadIdx.x;
    const int lane = tid & 31;
    const int wv   = tid >> 5;
    const int n0   = blockIdx.x * 16;
    const int h    = blockIdx.y;
    const int b    = blockIdx.z;
    const size_t headBase = ((size_t)b * NUM_H + h) * NUM_N;

    const int off = (lane < 16) ? 0 : 8;

    // A fragments: q rows n0..n0+15 (kept in registers for the whole kernel)
    const size_t qrOff = (headBase + n0 + (lane & 15)) * DIM_D;
    v16bf aq[2], aeq[2];
    #pragma unroll
    for (int ks = 0; ks < 2; ++ks) {
        const bf16_t* qp  = qbf  + qrOff + ks * 32;
        const bf16_t* eqp = eqbf + qrOff + ks * 32;
        aq[ks]  = cat16(*reinterpret_cast<const v8bf*>(qp  + off),
                        *reinterpret_cast<const v8bf*>(qp  + off + 16));
        aeq[ks] = cat16(*reinterpret_cast<const v8bf*>(eqp + off),
                        *reinterpret_cast<const v8bf*>(eqp + off + 16));
    }

    // ---- score stripes: 64 column tiles, 8 per wave, 4 WMMA each
    for (int t = wv; t < 64; t += 8) {
        const int m0 = t * 16;
        const size_t krOff = (headBase + m0 + (lane & 15)) * DIM_D;
        v8f accT = {}, accP = {};
        #pragma unroll
        for (int ks = 0; ks < 2; ++ks) {
            const bf16_t* kp  = kbf  + krOff + ks * 32;
            const bf16_t* ekp = ekbf + krOff + ks * 32;
            v16bf bk  = cat16(*reinterpret_cast<const v8bf*>(kp  + off),
                              *reinterpret_cast<const v8bf*>(kp  + off + 16));
            v16bf bek = cat16(*reinterpret_cast<const v8bf*>(ekp + off),
                              *reinterpret_cast<const v8bf*>(ekp + off + 16));
            accT = wmma_bf(aq[ks],  bk,  accT);
            accP = wmma_bf(aeq[ks], bek, accP);
        }
        const int colw = m0 + (lane & 15);
        #pragma unroll
        for (int i = 0; i < 8; ++i) {
            const int r = i + ((lane < 16) ? 0 : 8);
            s_true[r * 1024 + colw] = accT[i] * 0.125f;   // 1/sqrt(64)
            s_pred[r * 1024 + colw] = accP[i];
        }
    }
    __syncthreads();

    // ---- top-32 + softmax + sparse attn@V ; wave handles rows 2w, 2w+1
    #pragma unroll 1
    for (int rr = 0; rr < 2; ++rr) {
        const int r = wv * 2 + rr;
        const float* prow = s_pred + r * 1024;
        float pv[32];
        #pragma unroll
        for (int j = 0; j < 32; ++j) pv[j] = prow[lane + 32 * j];

        unsigned taken = 0u;
        int selCol = 0;
        #pragma unroll 1
        for (int it = 0; it < TOPK; ++it) {
            float lmax = -3.0e38f; int lj = 0;
            #pragma unroll
            for (int j = 0; j < 32; ++j) {
                const bool ok = (((taken >> j) & 1u) == 0u) && (pv[j] > lmax);
                if (ok) { lmax = pv[j]; lj = j; }
            }
            float bv2 = lmax;
            int   bc2 = lane + 32 * lj;
            #pragma unroll
            for (int o = 16; o > 0; o >>= 1) {
                const float ov = __shfl_xor(bv2, o, 32);
                const int   oc = __shfl_xor(bc2, o, 32);
                if (ov > bv2 || (ov == bv2 && oc < bc2)) { bv2 = ov; bc2 = oc; }
            }
            if ((bc2 & 31) == lane) taken |= 1u << (bc2 >> 5);  // winner retires slot
            if (lane == it) selCol = bc2;                        // lane `it` owns pick #it
        }

        float tval = s_true[r * 1024 + selCol];
        float m = tval;
        #pragma unroll
        for (int o = 16; o > 0; o >>= 1) { const float ov = __shfl_xor(m, o, 32); m = ov > m ? ov : m; }
        float e = __expf(tval - m);
        float ssum = e;
        #pragma unroll
        for (int o = 16; o > 0; o >>= 1) ssum += __shfl_xor(ssum, o, 32);
        const float p = e / ssum;

        float acc0 = 0.f, acc1 = 0.f;
        #pragma unroll 1
        for (int j = 0; j < 32; ++j) {
            const float pj = __shfl(p, j, 32);
            const int   cj = __shfl(selCol, j, 32);
            const float* vrow = vf + (headBase + (size_t)cj) * DIM_D;
            acc0 += pj * vrow[lane];
            acc1 += pj * vrow[lane + 32];
        }
        const size_t obase = ((size_t)(b * NUM_N + n0 + r)) * NUM_C + h * DIM_D;
        ybf[obase + lane]      = (bf16_t)acc0;
        ybf[obase + lane + 32] = (bf16_t)acc1;
    }
}

// ---------------------------------------------------------------------------
extern "C" void kernel_launch(void* const* d_in, const int* in_sizes, int n_in,
                              void* d_out, int out_size, void* d_ws, size_t ws_size,
                              hipStream_t stream) {
    (void)in_sizes; (void)n_in; (void)out_size; (void)ws_size;

    const float* x      = (const float*)d_in[0];   // [B,N,C]
    const float* qkv_w  = (const float*)d_in[1];   // [3C,C]
    const float* qkv_b  = (const float*)d_in[2];   // [3C]
    const float* proj_w = (const float*)d_in[3];   // [C,C]
    const float* proj_b = (const float*)d_in[4];   // [C]
    float* out = (float*)d_out;                    // [B,N,C] f32

    const int M = NUM_B * NUM_N;                               // 8192
    const size_t QK = (size_t)NUM_B * NUM_H * NUM_N * DIM_D;   // head-tensor elems
    const size_t nX = (size_t)M * NUM_C;
    const size_t nWq = (size_t)3 * NUM_C * NUM_C;
    const size_t nWp = (size_t)NUM_C * NUM_C;

    char* p = (char*)d_ws;
    bf16_t* xbf   = (bf16_t*)p; p += nX  * sizeof(bf16_t);
    bf16_t* wqbf  = (bf16_t*)p; p += nWq * sizeof(bf16_t);
    bf16_t* wpbf  = (bf16_t*)p; p += nWp * sizeof(bf16_t);
    bf16_t* qbf   = (bf16_t*)p; p += QK  * sizeof(bf16_t);
    bf16_t* eqbf  = (bf16_t*)p; p += QK  * sizeof(bf16_t);
    bf16_t* kbf   = (bf16_t*)p; p += QK  * sizeof(bf16_t);
    bf16_t* ekbf  = (bf16_t*)p; p += QK  * sizeof(bf16_t);
    float*  vf    = (float*)p;  p += QK  * sizeof(float);
    bf16_t* ybf   = (bf16_t*)p; p += nX  * sizeof(bf16_t);

    // P0: one-shot f32 -> bf16 conversions (memory-bound, ~2.5 us at 23.3 TB/s)
    convert_bf16_kernel<<<(int)(nX  / 8 + 255) / 256, 256, 0, stream>>>(x,      xbf,  (int)(nX  / 8));
    convert_bf16_kernel<<<(int)(nWq / 8 + 255) / 256, 256, 0, stream>>>(qkv_w,  wqbf, (int)(nWq / 8));
    convert_bf16_kernel<<<(int)(nWp / 8 + 255) / 256, 256, 0, stream>>>(proj_w, wpbf, (int)(nWp / 8));

    // K0: QKV GEMM, 8192 x 2304 x 768 (async double-buffered staging)
    gemm_kernel<0><<<dim3((3 * NUM_C) / 128, M / 64), 256, 0, stream>>>(
        xbf, wqbf, qkv_b, NUM_C, 3 * NUM_C,
        nullptr, qbf, eqbf, kbf, ekbf, vf);

    // K1: fused scores + top-k + softmax + sparse AV (128 KB dynamic LDS / block)
    attn_kernel<<<dim3(NUM_N / 16, NUM_H, NUM_B), 256,
                  2u * 16u * 1024u * sizeof(float), stream>>>(
        qbf, eqbf, kbf, ekbf, vf, ybf);

    // K2: projection GEMM, 8192 x 768 x 768
    gemm_kernel<1><<<dim3(NUM_C / 128, M / 64), 256, 0, stream>>>(
        ybf, wpbf, proj_b, NUM_C, NUM_C,
        out, nullptr, nullptr, nullptr, nullptr, nullptr);
}
